// GraphMatcher_89739046683485
// MI455X (gfx1250) — matrix operand order repaired and checked by reading, not compiled, not run on hardware
//
#include <hip/hip_runtime.h>

typedef _Float16 h16;
typedef __attribute__((ext_vector_type(16))) _Float16 v16h;
typedef __attribute__((ext_vector_type(8)))  _Float16 v8h;
typedef __attribute__((ext_vector_type(8)))  float    v8f;
typedef __attribute__((ext_vector_type(4)))  unsigned int u32x4;
typedef __attribute__((ext_vector_type(8)))  unsigned int u32x8;

#define DD 256
#define NN 1024
#define LL 18

// ---- WMMA operand helpers (layouts per CDNA5 ISA 7.12.2, wave32) ----
__device__ inline v16h cat16(v8h lo, v8h hi) {
  return __builtin_shufflevector(lo, hi, 0,1,2,3,4,5,6,7,8,9,10,11,12,13,14,15);
}
// A operand (16x32 f16): lane<16 -> K {0..7,16..23}; lane>=16 -> K {8..15,24..31}
// p must already point at (row_base + (lane>=16 ? 8 : 0))
__device__ inline v16h ldA(const h16* p) {
  v8h lo = *(const v8h*)(p);
  v8h hi = *(const v8h*)(p + 16);
  return cat16(lo, hi);
}
// B operand (32x16 f16): lane<16 -> K {0..7,16..23}; lane>=16 -> K {8..15,24..31}
// p must already point at (col_base + (lane>=16 ? 8 : 0))
__device__ inline v16h ldB(const h16* p) {
  v8h lo = *(const v8h*)(p);
  v8h hi = *(const v8h*)(p + 16);
  return cat16(lo, hi);
}
__device__ inline v8f wmma_f16(v16h a, v16h b, v8f c) {
  return __builtin_amdgcn_wmma_f32_16x16x32_f16(false, a, false, b, (short)0, c, false, false);
}
__device__ inline float redmax16(float v) {
  v = fmaxf(v, __shfl_xor(v, 1, 32));
  v = fmaxf(v, __shfl_xor(v, 2, 32));
  v = fmaxf(v, __shfl_xor(v, 4, 32));
  v = fmaxf(v, __shfl_xor(v, 8, 32));
  return v;
}
__device__ inline float redsum16(float v) {
  v += __shfl_xor(v, 1, 32);
  v += __shfl_xor(v, 2, 32);
  v += __shfl_xor(v, 4, 32);
  v += __shfl_xor(v, 8, 32);
  return v;
}

// ---- prep kernels ----
__global__ void f32tof16(const float* __restrict__ in, h16* __restrict__ out, int n) {
  int i = blockIdx.x * blockDim.x + threadIdx.x;
  if (i < n) out[i] = (h16)in[i];
}
__global__ void bnprep(const float* __restrict__ gamma, const float* __restrict__ beta,
                       const float* __restrict__ mu, const float* __restrict__ var,
                       const float* __restrict__ b1, float* __restrict__ sc,
                       float* __restrict__ sh, int n) {
  int i = blockIdx.x * blockDim.x + threadIdx.x;
  if (i < n) {
    float s = gamma[i] * rsqrtf(var[i] + 1e-5f);
    sc[i] = s;
    sh[i] = beta[i] + (b1[i] - mu[i]) * s;
  }
}
__global__ void initr(const float* __restrict__ d0, const float* __restrict__ d1,
                      float* __restrict__ r) {
  int i = blockIdx.x * blockDim.x + threadIdx.x;
  const int halfsz = 2 * DD * NN;
  if (i < halfsz) r[i] = d0[i];
  else if (i < 2 * halfsz) r[i] = d1[i - halfsz];
}
// y_t[bb][n][c] (c<256) = (h16) r[bb][c][n]
__global__ void buildx(const float* __restrict__ r, h16* __restrict__ y_t) {
  int i = blockIdx.x * blockDim.x + threadIdx.x;  // 4*1024*256
  int bb = i >> 18;
  int rem = i & 262143;
  int n = rem >> 8;
  int c = rem & 255;
  y_t[((size_t)(bb * NN + n)) * 512 + c] = (h16)r[((size_t)(bb * DD + c)) * NN + n];
}

// ---- generic 16x64-per-wave f16 GEMM with fused epilogues ----
// A (16 x K weight panel) is staged into LDS once per workgroup via the
// Tensor Data Mover (tensor_load_to_lds, TENSORcnt), then shared by all 8
// waves -> 8x reduction in redundant weight traffic.
// MODE 0: q/k proj  -> outh = headT [bb][h][n][d]   (f16)
// MODE 1: v proj    -> outh = headM [bb][h][d][m]   (f16)
// MODE 2: merge     -> outh = y_t[bb][col][256+row] (f16)
// MODE 3: MLP1+BN+ReLU -> outh = h_t[bb][col][row]  (f16)
// MODE 4: MLP2 + residual into racc (f32), optional copy to outf
template <int MODE>
__global__ __launch_bounds__(256) void gemm16(
    const h16* __restrict__ A, const float* __restrict__ bias,
    const h16* __restrict__ B, int K, int ldb, int swap,
    h16* __restrict__ outh, const float* __restrict__ sc,
    const float* __restrict__ sh, float* __restrict__ racc,
    float* __restrict__ outf) {
  __shared__ alignas(16) h16 shA[16 * 512];  // 16 rows x K(<=512) halfs
  const int wave = threadIdx.x >> 5;
  const int lane = threadIdx.x & 31;
  const int hl = lane >> 4, lc = lane & 15;
  const int row0 = blockIdx.y * 16;
  const int col0 = blockIdx.x * 512 + wave * 64;
  const int bb = blockIdx.z;
  const int sbb = swap ? (bb ^ 2) : bb;

  // --- TDM: DMA the contiguous 16xK weight panel (32*K bytes) into LDS ---
  if (threadIdx.x < 32) {
    unsigned long long ga = (unsigned long long)(size_t)(A + (size_t)row0 * K);
    unsigned int lds = (unsigned int)(size_t)(&shA[0]);
    unsigned int ndw = (unsigned int)(K * 8);  // panel size in dwords
    u32x4 g0;
    g0[0] = 1u;                                      // count=1, user mode
    g0[1] = lds;                                     // lds_addr (bytes)
    g0[2] = (unsigned int)ga;                        // global_addr lo
    g0[3] = (unsigned int)(ga >> 32) | 0x80000000u;  // addr hi | type=2
    u32x8 g1;
    g1[0] = 2u << 16;                  // data_size=4B, no mask/pad/iterate
    g1[1] = (ndw & 0xffffu) << 16;     // tensor_dim0[15:0]
    g1[2] = (ndw >> 16) | (1u << 16);  // tensor_dim0[31:16] | tensor_dim1=1
    g1[3] = ndw << 16;                 // tile_dim0 = ndw
    g1[4] = 1u;                        // tile_dim1=1, tile_dim2=0
    g1[5] = ndw;                       // tensor_dim0_stride lo
    g1[6] = 0u;
    g1[7] = 0u;
    asm volatile("tensor_load_to_lds %0, %1" :: "s"(g0), "s"(g1) : "memory");
    __builtin_amdgcn_s_wait_tensorcnt(0);
  }
  __syncthreads();

  const h16* ap = shA + lc * K + hl * 8;
  const h16* bbase = B + (size_t)sbb * NN * ldb;
  const h16* bp[4];
#pragma unroll
  for (int t = 0; t < 4; t++)
    bp[t] = bbase + (size_t)(col0 + t * 16 + lc) * ldb + hl * 8;

  v8f zero = {0.f, 0.f, 0.f, 0.f, 0.f, 0.f, 0.f, 0.f};
  v8f acc[4] = {zero, zero, zero, zero};

  for (int kk = 0; kk < K; kk += 32) {
    v16h a = ldA(ap + kk);
    if (kk + 32 < K) __builtin_prefetch(bp[0] + kk + 32, 0, 1);
#pragma unroll
    for (int t = 0; t < 4; t++) {
      v16h b = ldB(bp[t] + kk);
      acc[t] = wmma_f16(a, b, acc[t]);
    }
  }

#pragma unroll
  for (int j = 0; j < 8; j++) {
    const int row = row0 + j + hl * 8;
    float bv = 0.f;
    if constexpr (MODE != 3) bv = bias[row];
    float s0 = 0.f, s1 = 0.f;
    if constexpr (MODE == 3) { s0 = sc[row]; s1 = sh[row]; }
#pragma unroll
    for (int t = 0; t < 4; t++) {
      const int col = col0 + t * 16 + lc;
      float v = acc[t][j] + bv;
      if constexpr (MODE == 0) {
        int hh = row & 3, d = row >> 2;
        outh[(((size_t)bb * 4 + hh) * NN + col) * 64 + d] = (h16)v;
      } else if constexpr (MODE == 1) {
        int hh = row & 3, d = row >> 2;
        outh[(((size_t)bb * 4 + hh) * 64 + d) * NN + col] = (h16)v;
      } else if constexpr (MODE == 2) {
        outh[((size_t)bb * NN + col) * 512 + 256 + row] = (h16)v;
      } else if constexpr (MODE == 3) {
        float x = fmaxf(fmaf(acc[t][j], s0, s1), 0.f);
        outh[((size_t)bb * NN + col) * 512 + row] = (h16)x;
      } else {
        float* rp = racc + ((size_t)bb * DD + row) * NN + col;
        float nv = *rp + v;
        *rp = nv;
        if (outf) {
          int side = bb >> 1, bsub = bb & 1;
          outf[(size_t)side * 4718592 + (size_t)bsub * 262144 +
               (size_t)row * NN + col] = nv;
        }
      }
    }
  }
}

// ---- fused flash attention: per wave 16 query rows, all 64 head dims ----
// qT,kT: [bb*4+h][n][d]  (f16, d contiguous)   vM: [bb*4+h][d][m] (m contiguous)
// att_t: [bb][n][256]  with channel = d*4+h
__global__ __launch_bounds__(128) void attn_flash(
    const h16* __restrict__ qT, const h16* __restrict__ kT,
    const h16* __restrict__ vM, h16* __restrict__ att_t) {
  __shared__ alignas(16) h16 plds[4][512];  // per-wave 16x32 P tile
  const int wave = threadIdx.x >> 5;
  const int lane = threadIdx.x & 31;
  const int hl = lane >> 4, lc = lane & 15;
  const int comb = blockIdx.y;  // bb*4 + h
  const int n0 = blockIdx.x * 64 + wave * 16;

  const h16* q = qT + (size_t)comb * NN * 64;
  const h16* k = kT + (size_t)comb * NN * 64;
  const h16* v = vM + (size_t)comb * 64 * NN;

  const h16* qp = q + (size_t)(n0 + lc) * 64 + hl * 8;
  v16h a0 = ldA(qp);        // d 0..31
  v16h a1 = ldA(qp + 32);   // d 32..63

  v8f zero = {0.f, 0.f, 0.f, 0.f, 0.f, 0.f, 0.f, 0.f};
  v8f o[4] = {zero, zero, zero, zero};
  float rmax[8], rsum[8];
#pragma unroll
  for (int j = 0; j < 8; j++) { rmax[j] = -3.0e30f; rsum[j] = 0.f; }

  h16* pl = plds[wave];

  for (int m0 = 0; m0 < NN; m0 += 32) {
    // scores for 32 keys: two 16x16 tiles
    v8f s[2];
#pragma unroll
    for (int si = 0; si < 2; si++) {
      const h16* kp = k + (size_t)(m0 + si * 16 + lc) * 64 + hl * 8;
      v16h b0 = ldB(kp);
      v16h b1 = ldB(kp + 32);
      v8f z = zero;
      z = wmma_f16(a0, b0, z);
      z = wmma_f16(a1, b1, z);
      s[si] = z * 0.125f;  // 1/sqrt(64)
    }
    // online softmax per row (row = j + hl*8, spread over 16 lanes)
#pragma unroll
    for (int j = 0; j < 8; j++) {
      float tm = redmax16(fmaxf(s[0][j], s[1][j]));
      float nm = fmaxf(rmax[j], tm);
      float corr = __expf(rmax[j] - nm);
      rmax[j] = nm;
      float p0 = __expf(s[0][j] - nm);
      float p1 = __expf(s[1][j] - nm);
      rsum[j] = rsum[j] * corr + redsum16(p0 + p1);
      int rr = j + hl * 8;
      pl[rr * 32 + lc] = (h16)p0;
      pl[rr * 32 + 16 + lc] = (h16)p1;
#pragma unroll
      for (int t = 0; t < 4; t++) o[t][j] *= corr;
    }
    // P (16x32) as A operand from LDS, V (32x64) as B operand
    v16h apP = ldA(pl + lc * 32 + hl * 8);
#pragma unroll
    for (int t = 0; t < 4; t++) {
      const h16* vp = v + (size_t)(t * 16 + lc) * NN + m0 + hl * 8;
      v16h b = ldB(vp);
      o[t] = wmma_f16(apP, b, o[t]);
    }
  }

  const int bb = comb >> 2, hh = comb & 3;
  float inv[8];
#pragma unroll
  for (int j = 0; j < 8; j++) inv[j] = 1.0f / rsum[j];
#pragma unroll
  for (int t = 0; t < 4; t++) {
#pragma unroll
    for (int j = 0; j < 8; j++) {
      int n = n0 + j + hl * 8;
      int d = t * 16 + lc;
      att_t[((size_t)bb * NN + n) * DD + ((d << 2) | hh)] = (h16)(o[t][j] * inv[j]);
    }
  }
}

extern "C" void kernel_launch(void* const* d_in, const int* in_sizes, int n_in,
                              void* d_out, int out_size, void* d_ws, size_t ws_size,
                              hipStream_t stream) {
  const float* desc0 = (const float*)d_in[0];
  const float* desc1 = (const float*)d_in[1];
  const float* Wq = (const float*)d_in[2];
  const float* bq = (const float*)d_in[3];
  const float* Wk = (const float*)d_in[4];
  const float* bk = (const float*)d_in[5];
  const float* Wv = (const float*)d_in[6];
  const float* bv = (const float*)d_in[7];
  const float* Wm = (const float*)d_in[8];
  const float* bm = (const float*)d_in[9];
  const float* W1 = (const float*)d_in[10];
  const float* b1 = (const float*)d_in[11];
  const float* gamma = (const float*)d_in[12];
  const float* beta = (const float*)d_in[13];
  const float* mu = (const float*)d_in[14];
  const float* var = (const float*)d_in[15];
  const float* W2 = (const float*)d_in[16];
  const float* b2 = (const float*)d_in[17];
  float* out = (float*)d_out;

  char* base = (char*)d_ws;
  size_t off = 0;
  auto alloc = [&](size_t bytes) -> char* {
    char* p = base + off;
    off = (off + bytes + 255) & ~(size_t)255;
    return p;
  };
  const int nWqkv = LL * DD * DD;        // 1179648
  const int nW1 = LL * 512 * 512;        // 4718592
  const int nW2 = LL * DD * 512;         // 2359296
  h16* Wq_h = (h16*)alloc((size_t)nWqkv * 2);
  h16* Wk_h = (h16*)alloc((size_t)nWqkv * 2);
  h16* Wv_h = (h16*)alloc((size_t)nWqkv * 2);
  h16* Wm_h = (h16*)alloc((size_t)nWqkv * 2);
  h16* W1_h = (h16*)alloc((size_t)nW1 * 2);
  h16* W2_h = (h16*)alloc((size_t)nW2 * 2);
  float* sc = (float*)alloc((size_t)LL * 512 * 4);
  float* sh = (float*)alloc((size_t)LL * 512 * 4);
  float* r = (float*)alloc((size_t)4 * DD * NN * 4);
  h16* y_t = (h16*)alloc((size_t)4 * NN * 512 * 2);
  h16* h_t = (h16*)alloc((size_t)4 * NN * 512 * 2);
  h16* att_t = (h16*)alloc((size_t)4 * NN * DD * 2);
  h16* qT = (h16*)alloc((size_t)16 * NN * 64 * 2);
  h16* kT = (h16*)alloc((size_t)16 * NN * 64 * 2);
  h16* vM = (h16*)alloc((size_t)16 * 64 * NN * 2);

  // ---- prep ----
  f32tof16<<<(nWqkv + 255) / 256, 256, 0, stream>>>(Wq, Wq_h, nWqkv);
  f32tof16<<<(nWqkv + 255) / 256, 256, 0, stream>>>(Wk, Wk_h, nWqkv);
  f32tof16<<<(nWqkv + 255) / 256, 256, 0, stream>>>(Wv, Wv_h, nWqkv);
  f32tof16<<<(nWqkv + 255) / 256, 256, 0, stream>>>(Wm, Wm_h, nWqkv);
  f32tof16<<<(nW1 + 255) / 256, 256, 0, stream>>>(W1, W1_h, nW1);
  f32tof16<<<(nW2 + 255) / 256, 256, 0, stream>>>(W2, W2_h, nW2);
  bnprep<<<(LL * 512 + 255) / 256, 256, 0, stream>>>(gamma, beta, mu, var, b1,
                                                     sc, sh, LL * 512);
  initr<<<(4 * DD * NN + 255) / 256, 256, 0, stream>>>(desc0, desc1, r);

  const dim3 gP(2, 16, 4);   // M=256 GEMMs
  const dim3 gM1(2, 32, 4);  // M=512 GEMM
  for (int i = 0; i < LL; i++) {
    const int swap = (i & 1);  // even=self, odd=cross
    buildx<<<4096, 256, 0, stream>>>(r, y_t);
    gemm16<0><<<gP, 256, 0, stream>>>(Wq_h + (size_t)i * DD * DD, bq + i * DD,
                                      y_t, 256, 512, 0, qT, nullptr, nullptr,
                                      nullptr, nullptr);
    gemm16<0><<<gP, 256, 0, stream>>>(Wk_h + (size_t)i * DD * DD, bk + i * DD,
                                      y_t, 256, 512, swap, kT, nullptr, nullptr,
                                      nullptr, nullptr);
    gemm16<1><<<gP, 256, 0, stream>>>(Wv_h + (size_t)i * DD * DD, bv + i * DD,
                                      y_t, 256, 512, swap, vM, nullptr, nullptr,
                                      nullptr, nullptr);
    attn_flash<<<dim3(16, 16), 128, 0, stream>>>(qT, kT, vM, att_t);
    gemm16<2><<<gP, 256, 0, stream>>>(Wm_h + (size_t)i * DD * DD, bm + i * DD,
                                      att_t, 256, 256, 0, y_t, nullptr, nullptr,
                                      nullptr, nullptr);
    gemm16<3><<<gM1, 256, 0, stream>>>(W1_h + (size_t)i * 512 * 512, nullptr,
                                       y_t, 512, 512, 0, h_t, sc + i * 512,
                                       sh + i * 512, nullptr, nullptr);
    float* op = (i & 1) ? (out + (size_t)(i / 2) * 2 * DD * NN) : nullptr;
    gemm16<4><<<gP, 256, 0, stream>>>(W2_h + (size_t)i * DD * 512, b2 + i * DD,
                                      h_t, 512, 512, 0, nullptr, nullptr,
                                      nullptr, r, op);
  }
}
